// CSM_82575041233788
// MI455X (gfx1250) — matrix operand order repaired and checked by reading, not compiled, not run on hardware
//
#include <hip/hip_runtime.h>

// ---------------------------------------------------------------------------
// 3D Vision-Mamba block for MI455X (gfx1250, wave32, WMMA).
// Big GEMMs: double-buffered LDS tiles staged by global_load_async_to_lds_b128
// (ASYNCcnt-pipelined: tile t+1 DMA overlaps tile t WMMAs), feeding
// v_wmma_f32_16x16x32_f16. Edge GEMMs use guarded b128 staging.
// Scan: one wave32 per (batch, d_inner) channel, shfl_xor reduction.
// ---------------------------------------------------------------------------

#define TRY_ASYNC_LDS 1   // set 0 if the async-to-LDS asm is rejected

typedef _Float16 v16h __attribute__((ext_vector_type(16)));
typedef _Float16 v8h  __attribute__((ext_vector_type(8)));
typedef float    v8f  __attribute__((ext_vector_type(8)));
typedef __attribute__((address_space(3))) _Float16 lds_f16;

#define DM     768          // d_model
#define DI     1536         // d_inner
#define DSTATE 64
#define DTR    48
#define XPW    176          // dt_rank + 2*d_state
#define BSZ    2
#define LSEQ   2048         // 8*16*16
#define NT     (BSZ*LSEQ)   // 4096 tokens
#define NTOT   (NT*DM)      // 3,145,728
#define EPSF   1e-5f
#define SLOPEF 0.01f

// ---------------------------------------------------------------------------
// elementwise helpers
// ---------------------------------------------------------------------------
__global__ __launch_bounds__(256) void k_zero(float* p, int n) {
  int i = blockIdx.x * 256 + threadIdx.x;
  if (i < n) p[i] = 0.0f;
}

__global__ __launch_bounds__(256) void k_cvt16(const float* __restrict__ s,
                                               _Float16* __restrict__ d, int n) {
  int i = blockIdx.x * 256 + threadIdx.x;
  if (i < n) d[i] = (_Float16)s[i];
}

// BatchNorm3d (eval) + LeakyReLU, layout [B, C, D, H, W]
__global__ __launch_bounds__(256) void k_bn_lrelu(
    const float* __restrict__ x, const float* __restrict__ g,
    const float* __restrict__ b, const float* __restrict__ m,
    const float* __restrict__ v, float* __restrict__ o) {
  int i = blockIdx.x * 256 + threadIdx.x;
  if (i >= NTOT) return;
  int c = (i >> 11) % DM;                       // D*H*W = 2048
  float h = (x[i] - m[c]) * rsqrtf(v[c] + EPSF) * g[c] + b[c];
  o[i] = h > 0.0f ? h : SLOPEF * h;
}

// depthwise 3x3x3 SAME conv over [D=8, H=16, W=16]
__global__ __launch_bounds__(256) void k_dwconv3(const float* __restrict__ in,
                                                 const float* __restrict__ w,
                                                 float* __restrict__ out) {
  int i = blockIdx.x * 256 + threadIdx.x;
  if (i >= NTOT) return;
  int ww = i & 15, hh = (i >> 4) & 15, dd = (i >> 8) & 7;
  int c = (i >> 11) % DM, b = i / (2048 * DM);
  const float* src = in + ((size_t)(b * DM + c)) * 2048;
  const float* wt  = w + c * 27;
  float acc = 0.0f;
  #pragma unroll
  for (int kd = -1; kd <= 1; ++kd) {
    int zd = dd + kd; if (zd < 0 || zd >= 8) continue;
    #pragma unroll
    for (int kh = -1; kh <= 1; ++kh) {
      int zh = hh + kh; if (zh < 0 || zh >= 16) continue;
      #pragma unroll
      for (int kw = -1; kw <= 1; ++kw) {
        int zw = ww + kw; if (zw < 0 || zw >= 16) continue;
        acc += src[zd * 256 + zh * 16 + zw] * wt[(kd+1)*9 + (kh+1)*3 + (kw+1)];
      }
    }
  }
  out[i] = acc;
}

// [B, C, L] f32 -> [B*L, C] f16 (token-major for WMMA A operand)
__global__ __launch_bounds__(256) void k_pack_tok(const float* __restrict__ in,
                                                  _Float16* __restrict__ out) {
  int i = blockIdx.x * 256 + threadIdx.x;
  if (i >= NTOT) return;
  int c = i % DM, t = i / DM;       // t = b*L + l
  int b = t >> 11, l = t & 2047;
  out[i] = (_Float16)in[((size_t)(b * DM + c)) * 2048 + l];
}

// LayerNorm over channels (768) per token; emits f16 activations
__global__ __launch_bounds__(256) void k_layernorm(
    const float* __restrict__ h, const float* __restrict__ g,
    const float* __restrict__ b, _Float16* __restrict__ o) {
  __shared__ float rs[256], rss[256];
  int tok = blockIdx.x, t = threadIdx.x;
  float s = 0.0f, ss = 0.0f, val[3];
  #pragma unroll
  for (int j = 0; j < 3; ++j) {
    val[j] = h[(size_t)tok * DM + t + j * 256];
    s += val[j]; ss += val[j] * val[j];
  }
  rs[t] = s; rss[t] = ss; __syncthreads();
  for (int off = 128; off > 0; off >>= 1) {
    if (t < off) { rs[t] += rs[t + off]; rss[t] += rss[t + off]; }
    __syncthreads();
  }
  float mu = rs[0] * (1.0f / DM);
  float var = rss[0] * (1.0f / DM) - mu * mu;
  float inv = rsqrtf(var + EPSF);
  #pragma unroll
  for (int j = 0; j < 3; ++j) {
    int c = t + j * 256;
    o[(size_t)tok * DM + c] = (_Float16)((val[j] - mu) * inv * g[c] + b[c]);
  }
}

// directional flip of [B, L, C] f16 activations
__global__ __launch_bounds__(256) void k_flip(const _Float16* __restrict__ in,
                                              _Float16* __restrict__ out,
                                              int flipL, int flipC) {
  int i = blockIdx.x * 256 + threadIdx.x;
  if (i >= NTOT) return;
  int c = i % DM, t = i / DM, b = t >> 11, l = t & 2047;
  int sl = flipL ? (LSEQ - 1 - l) : l;
  int sc = flipC ? (DM - 1 - c) : c;
  out[i] = in[((size_t)(b * LSEQ + sl)) * DM + sc];
}

// causal depthwise conv1d (width 4) + bias + SiLU over xz[:, 0:DI]
__global__ __launch_bounds__(256) void k_conv1d_silu(
    const float* __restrict__ xz, const float* __restrict__ cw,
    const float* __restrict__ cb, float* __restrict__ xm,
    _Float16* __restrict__ xm16) {
  int i = blockIdx.x * 256 + threadIdx.x;
  if (i >= NT * DI) return;
  int d = i % DI, tok = i / DI, b = tok >> 11, l = tok & 2047;
  float acc = cb[d];
  #pragma unroll
  for (int k = 0; k < 4; ++k) {
    int ll = l - 3 + k;
    if (ll >= 0) acc += xz[((size_t)(b * LSEQ + ll)) * (2 * DI) + d] * cw[d * 4 + k];
  }
  float s = acc / (1.0f + __expf(-acc));      // SiLU
  xm[i] = s;
  xm16[i] = (_Float16)s;
}

// extract first DTR columns of xp as f16 (A operand of the dt GEMM)
__global__ __launch_bounds__(256) void k_pack_dt(const float* __restrict__ xp,
                                                 _Float16* __restrict__ o) {
  int i = blockIdx.x * 256 + threadIdx.x;
  if (i >= NT * DTR) return;
  int r = i % DTR, tok = i / DTR;
  o[i] = (_Float16)xp[(size_t)tok * XPW + r];
}

// ---------------------------------------------------------------------------
// Selective scan: one wave32 per (b, d_inner) channel; 2 states per lane.
// ---------------------------------------------------------------------------
__global__ __launch_bounds__(256) void k_scan(
    const float* __restrict__ dt, const float* __restrict__ xm,
    const float* __restrict__ xp, const float* __restrict__ xz,
    const float* __restrict__ A_log, const float* __restrict__ Dsk,
    _Float16* __restrict__ y16) {
  int w = blockIdx.x * 8 + (threadIdx.x >> 5);  // 3072 waves total
  int lane = threadIdx.x & 31;
  int b = w / DI, d = w % DI;
  int n0 = lane, n1 = lane + 32;
  float A0 = -__expf(A_log[d * DSTATE + n0]);
  float A1 = -__expf(A_log[d * DSTATE + n1]);
  float Dd = Dsk[d];
  float h0 = 0.0f, h1 = 0.0f;
  for (int l = 0; l < LSEQ; ++l) {
    int tok = b * LSEQ + l;
    float dtv = dt[(size_t)tok * DI + d];           // broadcast load
    float xv  = xm[(size_t)tok * DI + d];
    float Bv0 = xp[(size_t)tok * XPW + DTR + n0];
    float Bv1 = xp[(size_t)tok * XPW + DTR + n1];
    float Cv0 = xp[(size_t)tok * XPW + DTR + DSTATE + n0];
    float Cv1 = xp[(size_t)tok * XPW + DTR + DSTATE + n1];
    float dtx = dtv * xv;
    h0 = __expf(dtv * A0) * h0 + dtx * Bv0;
    h1 = __expf(dtv * A1) * h1 + dtx * Bv1;
    float y = h0 * Cv0 + h1 * Cv1;
    #pragma unroll
    for (int off = 16; off >= 1; off >>= 1) y += __shfl_xor(y, off, 32);
    if (lane == 0) {
      float z = xz[(size_t)tok * (2 * DI) + DI + d];
      float out = (y + xv * Dd) * (z / (1.0f + __expf(-z)));
      y16[(size_t)tok * DI + d] = (_Float16)out;
    }
  }
}

// ---------------------------------------------------------------------------
// WMMA GEMM (NT): C[M,N] = A[M,K] * B[N,K]^T, f16 in, f32 out.
// 64x64 tile per block (4 waves, each 16x64), K tile = 64.
// FULL=1: M,N,K multiples of 64 -> branch-free; tiles double-buffered via
//   global_load_async_to_lds_b128 (8 async issues/wave/tile; s_wait_asynccnt 8
//   drains only the in-flight previous tile -> DMA overlaps WMMA).
// FULL=0: chunk-guarded b128 staging (N=176 / K=48 cases).
// ---------------------------------------------------------------------------
#define EP_PLAIN    0
#define EP_LRELU    1
#define EP_SOFTPLUS 2
#define EP_FLIPACC  3

template <int MODE, int FULL>
__global__ __launch_bounds__(128) void k_gemm_wmma(
    const _Float16* __restrict__ A, const _Float16* __restrict__ Bw,
    float* __restrict__ C, int M, int N, int K,
    const float* __restrict__ bias, int flipL, int flipC) {
  __shared__ alignas(16) _Float16 sA[2][64][72];
  __shared__ alignas(16) _Float16 sB[2][64][72];
  const int tid = threadIdx.x;
  const int wave = tid >> 5, lane = tid & 31;
  const int rowBlk = blockIdx.y * 64, colBlk = blockIdx.x * 64;

  v8f acc[4] = {};

  const int lr  = tid >> 1;            // 0..63: tile row this thread stages
  const int lcB = (tid & 1) * 32;      // 32-half half-row
  const int grow = rowBlk + lr;
  const int nrow = colBlk + lr;
  const _Float16* Arow = A  + (size_t)grow * K;
  const _Float16* Brow = Bw + (size_t)nrow * K;

  const int am   = (wave << 4) + (lane & 15);
  const int koff = (lane >> 4) << 3;   // 0 or 8
  const int kb   = (lane >> 4) << 4;   // 0 or 16

  // per-lane async DMA: 16B global -> LDS, tracked by ASYNCcnt (8 per tile)
  auto stageAsync = [&](int k0, int buf) {
#if TRY_ASYNC_LDS
    #pragma unroll
    for (int c = 0; c < 32; c += 8) {
      lds_f16* la = (lds_f16*)&sA[buf][lr][lcB + c];
      lds_f16* lb = (lds_f16*)&sB[buf][lr][lcB + c];
      const _Float16* ga = Arow + k0 + lcB + c;
      const _Float16* gb = Brow + k0 + lcB + c;
      asm volatile("global_load_async_to_lds_b128 %0, %1, off"
                   :: "v"(la), "v"(ga) : "memory");
      asm volatile("global_load_async_to_lds_b128 %0, %1, off"
                   :: "v"(lb), "v"(gb) : "memory");
    }
#else
    #pragma unroll
    for (int c = 0; c < 32; c += 8) {
      *reinterpret_cast<uint4*>(&sA[buf][lr][lcB + c]) =
          *reinterpret_cast<const uint4*>(Arow + k0 + lcB + c);
      *reinterpret_cast<uint4*>(&sB[buf][lr][lcB + c]) =
          *reinterpret_cast<const uint4*>(Brow + k0 + lcB + c);
    }
#endif
  };

  // 8 WMMAs over one 64-deep K tile
  auto computeTile = [&](int buf) {
    #pragma unroll
    for (int ks = 0; ks < 64; ks += 32) {
      v8h alo = *reinterpret_cast<const v8h*>(&sA[buf][am][ks + koff]);
      v8h ahi = *reinterpret_cast<const v8h*>(&sA[buf][am][ks + koff + 16]);
      v16h af = __builtin_shufflevector(alo, ahi,
                  0,1,2,3,4,5,6,7,8,9,10,11,12,13,14,15);
      v16h bf[4];
      #pragma unroll
      for (int nb = 0; nb < 4; ++nb) {
        const int bn = (nb << 4) + (lane & 15);
        v8h blo = *reinterpret_cast<const v8h*>(&sB[buf][bn][ks + koff]);
        v8h bhi = *reinterpret_cast<const v8h*>(&sB[buf][bn][ks + koff + 16]);
        bf[nb] = __builtin_shufflevector(blo, bhi,
                   0,1,2,3,4,5,6,7,8,9,10,11,12,13,14,15);
      }
      #pragma unroll
      for (int nb = 0; nb < 4; ++nb) {
        acc[nb] = __builtin_amdgcn_wmma_f32_16x16x32_f16(
            false, af, false, bf[nb], (short)0, acc[nb], false, false);
      }
    }
  };

  if (FULL) {
    const int nTiles = K >> 6;
    stageAsync(0, 0);
    for (int t = 0; t < nTiles; ++t) {
      const int cur = t & 1;
      if (t + 1 < nTiles) {
        stageAsync((t + 1) << 6, cur ^ 1);   // next tile DMA in flight
#if TRY_ASYNC_LDS
        asm volatile("s_wait_asynccnt 0x8" ::: "memory");  // drain tile t only
#endif
      } else {
#if TRY_ASYNC_LDS
        asm volatile("s_wait_asynccnt 0x0" ::: "memory");
#endif
      }
      __syncthreads();       // tile t visible to all waves
      computeTile(cur);
      __syncthreads();       // all reads of buf(cur) done before reuse
    }
  } else {
    for (int k0 = 0; k0 < K; k0 += 64) {
      #pragma unroll
      for (int c = 0; c < 32; c += 8) {
        int k = k0 + lcB + c;
        uint4 va = make_uint4(0u, 0u, 0u, 0u);
        uint4 vb = make_uint4(0u, 0u, 0u, 0u);
        bool kin = (k + 8) <= K;
        if (grow < M && kin) va = *reinterpret_cast<const uint4*>(Arow + k);
        if (nrow < N && kin) vb = *reinterpret_cast<const uint4*>(Brow + k);
        *reinterpret_cast<uint4*>(&sA[0][lr][lcB + c]) = va;
        *reinterpret_cast<uint4*>(&sB[0][lr][lcB + c]) = vb;
      }
      __syncthreads();
      if (k0 + 64 < K) {     // speculative prefetch (global_prefetch_b8)
        __builtin_prefetch(Arow + k0 + 64 + lcB, 0, 1);
        __builtin_prefetch(Brow + k0 + 64 + lcB, 0, 1);
      }
      computeTile(0);
      __syncthreads();
    }
  }

  // epilogue
  const int mbase = (wave << 4) + ((lane >> 4) << 3);
  const int ncol  = lane & 15;
  #pragma unroll
  for (int nb = 0; nb < 4; ++nb) {
    #pragma unroll
    for (int r = 0; r < 8; ++r) {
      int gm = rowBlk + mbase + r;
      int gn = colBlk + (nb << 4) + ncol;
      if (!FULL && (gm >= M || gn >= N)) continue;
      float v = acc[nb][r];
      if (MODE == EP_PLAIN) {
        C[(size_t)gm * N + gn] = v;
      } else if (MODE == EP_LRELU) {
        C[(size_t)gm * N + gn] = v > 0.0f ? v : SLOPEF * v;
      } else if (MODE == EP_SOFTPLUS) {
        v += bias[gn];
        C[(size_t)gm * N + gn] = v > 20.0f ? v : log1pf(__expf(v));
      } else {  // EP_FLIPACC: flipped write-accumulate into d_out, /4
        int b = gm >> 11, l = gm & 2047;
        int dl = flipL ? (LSEQ - 1 - l) : l;
        int dn = flipC ? (N - 1 - gn) : gn;
        C[((size_t)(b * LSEQ + dl)) * N + dn] += 0.25f * v;
      }
    }
  }
}

// ---------------------------------------------------------------------------
// host side
// ---------------------------------------------------------------------------
static inline int cdiv(int a, int b) { return (a + b - 1) / b; }

extern "C" void kernel_launch(void* const* d_in, const int* in_sizes, int n_in,
                              void* d_out, int out_size, void* d_ws, size_t ws_size,
                              hipStream_t stream) {
  const float* x        = (const float*)d_in[0];
  const float* bn_gamma = (const float*)d_in[1];
  const float* bn_beta  = (const float*)d_in[2];
  const float* bn_mean  = (const float*)d_in[3];
  const float* bn_var   = (const float*)d_in[4];
  const float* dw_w     = (const float*)d_in[5];
  const float* pw_w     = (const float*)d_in[6];
  const float* ln_gamma = (const float*)d_in[7];
  const float* ln_beta  = (const float*)d_in[8];
  const float* W_in     = (const float*)d_in[9];
  const float* conv_w   = (const float*)d_in[10];
  const float* conv_b   = (const float*)d_in[11];
  const float* W_x      = (const float*)d_in[12];
  const float* W_dt     = (const float*)d_in[13];
  const float* b_dt     = (const float*)d_in[14];
  const float* A_log    = (const float*)d_in[15];
  const float* D_skip   = (const float*)d_in[16];
  const float* W_out    = (const float*)d_in[17];
  float* out = (float*)d_out;

  // workspace carve
  size_t off = 0;
  char* base = (char*)d_ws;
  auto alloc = [&](size_t bytes) -> void* {
    void* p = base + off;
    off += (bytes + 255) & ~(size_t)255;
    return p;
  };
  float*    h0     = (float*)   alloc((size_t)NTOT * 4);
  float*    h1     = (float*)   alloc((size_t)NTOT * 4);
  _Float16* Apack  = (_Float16*)alloc((size_t)NTOT * 2);
  float*    h2     = (float*)   alloc((size_t)NTOT * 4);
  _Float16* xf16   = (_Float16*)alloc((size_t)NTOT * 2);
  _Float16* xdir16 = (_Float16*)alloc((size_t)NTOT * 2);
  _Float16* pw16   = (_Float16*)alloc((size_t)DM * DM * 2);
  _Float16* Win16  = (_Float16*)alloc((size_t)2 * DI * DM * 2);
  _Float16* Wx16   = (_Float16*)alloc((size_t)XPW * DI * 2);
  _Float16* Wdt16  = (_Float16*)alloc((size_t)DI * DTR * 2);
  _Float16* Wout16 = (_Float16*)alloc((size_t)DM * DI * 2);
  float*    xz     = (float*)   alloc((size_t)NT * 2 * DI * 4);
  float*    xm     = (float*)   alloc((size_t)NT * DI * 4);
  _Float16* xm16   = (_Float16*)alloc((size_t)NT * DI * 2);
  float*    xp     = (float*)   alloc((size_t)NT * XPW * 4);
  _Float16* dt16   = (_Float16*)alloc((size_t)NT * DTR * 2);
  float*    dtb    = (float*)   alloc((size_t)NT * DI * 4);
  _Float16* y16    = (_Float16*)alloc((size_t)NT * DI * 2);
  if (off > ws_size) return;  // workspace too small; leave output untouched

  const int TPB = 256;
  // 1) zero output accumulator
  k_zero<<<cdiv(out_size, TPB), TPB, 0, stream>>>(out, out_size);
  // 2) BN + LeakyReLU
  k_bn_lrelu<<<cdiv(NTOT, TPB), TPB, 0, stream>>>(x, bn_gamma, bn_beta, bn_mean, bn_var, h0);
  // 3) depthwise 3x3x3
  k_dwconv3<<<cdiv(NTOT, TPB), TPB, 0, stream>>>(h0, dw_w, h1);
  // 4) token-major f16 pack
  k_pack_tok<<<cdiv(NTOT, TPB), TPB, 0, stream>>>(h1, Apack);
  // 5) weight conversions
  k_cvt16<<<cdiv(DM * DM, TPB), TPB, 0, stream>>>(pw_w, pw16, DM * DM);
  k_cvt16<<<cdiv(2 * DI * DM, TPB), TPB, 0, stream>>>(W_in, Win16, 2 * DI * DM);
  k_cvt16<<<cdiv(XPW * DI, TPB), TPB, 0, stream>>>(W_x, Wx16, XPW * DI);
  k_cvt16<<<cdiv(DI * DTR, TPB), TPB, 0, stream>>>(W_dt, Wdt16, DI * DTR);
  k_cvt16<<<cdiv(DM * DI, TPB), TPB, 0, stream>>>(W_out, Wout16, DM * DI);
  // 6) pointwise conv as GEMM + LeakyReLU (full tiles)
  {
    dim3 g(DM / 64, NT / 64);
    k_gemm_wmma<EP_LRELU, 1><<<g, 128, 0, stream>>>(Apack, pw16, h2, NT, DM, DM, nullptr, 0, 0);
  }
  // 7) LayerNorm -> f16 activations
  k_layernorm<<<NT, 256, 0, stream>>>(h2, ln_gamma, ln_beta, xf16);

  // 8) four directional Mamba passes
  for (int dir = 0; dir < 4; ++dir) {
    int flipC = dir & 1;        // dir1, dir3: channel flip
    int flipL = (dir >> 1) & 1; // dir2, dir3: sequence flip
    k_flip<<<cdiv(NTOT, TPB), TPB, 0, stream>>>(xf16, xdir16, flipL, flipC);
    {   // input projection: [4096, 3072] = xdir @ W_in^T (full tiles)
      dim3 g((2 * DI) / 64, NT / 64);
      k_gemm_wmma<EP_PLAIN, 1><<<g, 128, 0, stream>>>(xdir16, Win16, xz, NT, 2 * DI, DM, nullptr, 0, 0);
    }
    k_conv1d_silu<<<cdiv(NT * DI, TPB), TPB, 0, stream>>>(xz, conv_w, conv_b, xm, xm16);
    {   // x-projection: [4096, 176] = xm @ W_x^T (N edge -> guarded)
      dim3 g(cdiv(XPW, 64), NT / 64);
      k_gemm_wmma<EP_PLAIN, 0><<<g, 128, 0, stream>>>(xm16, Wx16, xp, NT, XPW, DI, nullptr, 0, 0);
    }
    k_pack_dt<<<cdiv(NT * DTR, TPB), TPB, 0, stream>>>(xp, dt16);
    {   // dt: softplus(dt_lowrank @ W_dt^T + b_dt) (K=48 -> guarded)
      dim3 g(DI / 64, NT / 64);
      k_gemm_wmma<EP_SOFTPLUS, 0><<<g, 128, 0, stream>>>(dt16, Wdt16, dtb, NT, DI, DTR, b_dt, 0, 0);
    }
    // selective scan + gating -> y16
    k_scan<<<(BSZ * DI) / 8, 256, 0, stream>>>(dtb, xm, xp, xz, A_log, D_skip, y16);
    {   // output projection, flipped accumulate into d_out (full tiles)
      dim3 g(DM / 64, NT / 64);
      k_gemm_wmma<EP_FLIPACC, 1><<<g, 128, 0, stream>>>(y16, Wout16, out, NT, DM, DI, nullptr, flipL, flipC);
    }
  }
}
